// Codebook_49606872269101
// MI455X (gfx1250) — matrix operand order repaired and checked by reading, not compiled, not run on hardware
//
#include <hip/hip_runtime.h>
#include <hip/hip_bf16.h>

// ---- types matching gfx1250 WMMA builtin signatures ----
typedef __attribute__((ext_vector_type(16))) __bf16 v16bf;
typedef __attribute__((ext_vector_type(8)))  __bf16 v8bf;
typedef __attribute__((ext_vector_type(8)))  float  v8f;

#define NUM_CODES  2048
#define LATENT_DIM 256
#define NPIX       32768          // 32 * 32 * 32
#define ZQ_ELEMS   8388608        // 32 * 256 * 32 * 32
#define LDS_STRIDE 264            // 256 + 8 pad, keeps 16B alignment per row
#define TILE_PIX   128

// ---------------------------------------------------------------------------
// Kernel 1: codebook -> bf16 copy, per-code squared norm, zero loss accum.
// grid = 2048 (one block per code), block = 256 (one thread per dim)
// ---------------------------------------------------------------------------
__global__ __launch_bounds__(256) void vq_prep(const float* __restrict__ cb,
                                               __bf16* __restrict__ cb_bf,
                                               float* __restrict__ cbn,
                                               float* __restrict__ accp) {
    const int r = blockIdx.x;
    const int t = threadIdx.x;
    float v = cb[r * LATENT_DIM + t];
    cb_bf[r * LATENT_DIM + t] = (__bf16)v;
    float s = v * v;
    #pragma unroll
    for (int m = 16; m >= 1; m >>= 1) s += __shfl_xor(s, m, 32);
    __shared__ float ss[8];
    if ((t & 31) == 0) ss[t >> 5] = s;
    __syncthreads();
    if (t == 0) {
        float tot = 0.f;
        #pragma unroll
        for (int w = 0; w < 8; ++w) tot += ss[w];
        cbn[r] = tot;
        if (r == 0) accp[0] = 0.f;   // re-zero every launch (graph replays)
    }
}

// ---------------------------------------------------------------------------
// Kernel 2: argmin over codes via bf16 WMMA.
// grid = 256 (128 pixels each), block = 256 (8 waves x 16 pixel rows)
// dynamic LDS = TILE_PIX * LDS_STRIDE * 2 bytes
// ---------------------------------------------------------------------------
__global__ __launch_bounds__(256) void vq_argmin(const float* __restrict__ z,
                                                 const __bf16* __restrict__ cb_bf,
                                                 const float* __restrict__ cbn,
                                                 int* __restrict__ idxw,
                                                 float* __restrict__ out_idx) {
    extern __shared__ char dynsmem[];
    __bf16* zbf = (__bf16*)dynsmem;

    const int tile = blockIdx.x;
    const int n0   = tile * TILE_PIX;       // global pixel base
    const int b    = n0 >> 10;              // 1024 pixels per batch image
    const int hw0  = n0 & 1023;
    const int t    = threadIdx.x;
    const float* zb = z + (size_t)b * (LATENT_DIM * 1024) + hw0;

    // Stage z tile into LDS as bf16, pixel-major [p][c] with padded stride.
    // Coalesced: 128 contiguous floats per channel row.
    for (int i = t; i < LATENT_DIM * TILE_PIX; i += 256) {
        int c = i >> 7;          // / TILE_PIX
        int p = i & (TILE_PIX - 1);
        zbf[p * LDS_STRIDE + c] = (__bf16)zb[(size_t)c * 1024 + p];
    }
    __syncthreads();

    const int lane = t & 31;
    const int wave = t >> 5;
    const int mrow = lane & 15;   // M (pixel) for A, N (code) for B/D columns
    const int hh   = lane >> 4;   // half-wave selector
    const int p    = wave * 16 + mrow;

    // Preload all 8 A-fragments (16x32 bf16 each) from LDS into registers.
    // Layout: lane half hh element i holds K = hh*16 + i (contiguous, matches B).
    v16bf afrag[8];
    #pragma unroll
    for (int s = 0; s < 8; ++s) {
        const __bf16* base = zbf + p * LDS_STRIDE + s * 32 + hh * 16;
        v8bf lo = *(const v8bf*)(base);        // K = s*32 + hh*16 + 0..7
        v8bf hi = *(const v8bf*)(base + 8);    // K = s*32 + hh*16 + 8..15
        #pragma unroll
        for (int i = 0; i < 8; ++i) { afrag[s][i] = lo[i]; afrag[s][i + 8] = hi[i]; }
    }

    float best[8];
    int   bcode[8];
    #pragma unroll
    for (int j = 0; j < 8; ++j) { best[j] = 3.4e38f; bcode[j] = 0; }

    // Sweep all codes, 16 per chunk; this lane owns code column kc + mrow.
    for (int kc = 0; kc < NUM_CODES; kc += 16) {
        const int k = kc + mrow;
        const __bf16* brow = cb_bf + (size_t)k * LATENT_DIM + hh * 16;
        v8f acc{};
        #pragma unroll
        for (int s = 0; s < 8; ++s) {
            // B fragment: 16 contiguous K-values per lane (32B aligned).
            v16bf bf = *(const v16bf*)(brow + s * 32);
            acc = __builtin_amdgcn_wmma_f32_16x16x32_bf16(
                false, afrag[s], false, bf, (short)0, acc, false, false);
        }
        const float nrm = cbn[k];
        #pragma unroll
        for (int j = 0; j < 8; ++j) {
            // argmin score: ||e||^2 - 2 z.e   (||z||^2 is row-constant)
            float d = __builtin_fmaf(-2.0f, acc[j], nrm);
            if (d < best[j]) { best[j] = d; bcode[j] = k; }   // first-min ties
        }
    }

    // Reduce across the 16 lanes sharing each D row; lanes 0-15 hold M=j,
    // lanes 16-31 hold M=j+8 (per 16x16 f32 C/D layout).
    #pragma unroll
    for (int j = 0; j < 8; ++j) {
        float d = best[j];
        int   bi = bcode[j];
        #pragma unroll
        for (int mask = 8; mask >= 1; mask >>= 1) {
            float od = __shfl_xor(d, mask, 32);
            int   oi = __shfl_xor(bi, mask, 32);
            if (od < d || (od == d && oi < bi)) { d = od; bi = oi; }
        }
        if (mrow == 0) {
            int n = n0 + wave * 16 + j + hh * 8;
            idxw[n] = bi;
            out_idx[n] = (float)bi;    // idx output region of d_out
        }
    }
}

// ---------------------------------------------------------------------------
// Kernel 3: gather z_q (f32, NCHW) + accumulate sum((z_q - z)^2).
// grid = 512 (64 pixels each), block = 256 (64 pixels x 4 channel groups)
// ---------------------------------------------------------------------------
__global__ __launch_bounds__(256) void vq_gather(const float* __restrict__ z,
                                                 const float* __restrict__ cb,
                                                 const int* __restrict__ idxw,
                                                 float* __restrict__ zq_out,
                                                 float* __restrict__ accp) {
    const int n0  = blockIdx.x * 64;
    const int b   = n0 >> 10;
    const int hw0 = n0 & 1023;
    const int t   = threadIdx.x;

    __shared__ int sidx[64];
    if (t < 64) sidx[t] = idxw[n0 + t];
    __syncthreads();

    const int pw = t & 63;       // pixel within tile
    const int cg = t >> 6;       // channel group 0..3
    const float* cbr = cb + (size_t)sidx[pw] * LATENT_DIM;   // L2-resident gather
    const size_t basez = (size_t)b * (LATENT_DIM * 1024) + hw0 + pw;

    float sum = 0.f;
    #pragma unroll 4
    for (int c = cg; c < LATENT_DIM; c += 4) {
        float e = cbr[c];
        size_t pos = basez + (size_t)c * 1024;
        float zv = z[pos];
        zq_out[pos] = e;          // coalesced over hw
        float dlt = e - zv;
        sum = fmaf(dlt, dlt, sum);
    }
    #pragma unroll
    for (int m = 16; m >= 1; m >>= 1) sum += __shfl_xor(sum, m, 32);
    __shared__ float ss[8];
    if ((t & 31) == 0) ss[t >> 5] = sum;
    __syncthreads();
    if (t == 0) {
        float bs = 0.f;
        #pragma unroll
        for (int w = 0; w < 8; ++w) bs += ss[w];
        atomicAdd(accp, bs);
    }
}

// ---------------------------------------------------------------------------
// Kernel 4: loss = (1 + BETA) * SSE / count
// ---------------------------------------------------------------------------
__global__ void vq_finalize(const float* __restrict__ accp,
                            float* __restrict__ out_loss) {
    out_loss[0] = accp[0] * (1.25f / 8388608.0f);
}

// ---------------------------------------------------------------------------
extern "C" void kernel_launch(void* const* d_in, const int* in_sizes, int n_in,
                              void* d_out, int out_size, void* d_ws, size_t ws_size,
                              hipStream_t stream) {
    (void)in_sizes; (void)n_in; (void)out_size; (void)ws_size;
    const float* z  = (const float*)d_in[0];   // [32,256,32,32]
    const float* cb = (const float*)d_in[1];   // [2048,256]
    float* out = (float*)d_out;                // z_q (8388608) | idx (32768) | loss (1)

    char* ws = (char*)d_ws;
    __bf16* cb_bf = (__bf16*)ws;                                     // 1 MB
    float*  cbn   = (float*)(ws + (1u << 20));                       // 8 KB
    int*    idxw  = (int*)(ws + (1u << 20) + 8192);                  // 128 KB
    float*  accp  = (float*)(ws + (1u << 20) + 8192 + 131072);       // 4 B

    vq_prep<<<NUM_CODES, 256, 0, stream>>>(cb, cb_bf, cbn, accp);

    const size_t lds_bytes = (size_t)TILE_PIX * LDS_STRIDE * sizeof(__bf16); // 67.6 KB
    vq_argmin<<<NPIX / TILE_PIX, 256, lds_bytes, stream>>>(
        z, cb_bf, cbn, idxw, out + ZQ_ELEMS);

    vq_gather<<<NPIX / 64, 256, 0, stream>>>(z, cb, idxw, out, accp);

    vq_finalize<<<1, 1, 0, stream>>>(accp, out + ZQ_ELEMS + NPIX);
}
